// NestedNarx_62646392980119
// MI455X (gfx1250) — matrix operand order repaired and checked by reading, not compiled, not run on hardware
//
#include <hip/hip_runtime.h>

typedef __attribute__((ext_vector_type(16))) _Float16 v16h;
typedef __attribute__((ext_vector_type(8)))  float    v8f;

#define NTI 4096
#define NGR 1024
#define NXC 8
#define HID 64
#define INF 32
#define MAXD 3
#define TILES_PER_T (NGR / 16)                    // 64 tiles per time step
#define TILES ((NTI - MAXD) * TILES_PER_T)        // 4093 * 64 = 261952

union HFrag { v16h v; _Float16 h[16]; unsigned int u[8]; };
union FAcc  { v8f v; float f[8]; };

__device__ __forceinline__ float fast_tanh(float v) {
#if __has_builtin(__builtin_amdgcn_tanhf)
    return __builtin_amdgcn_tanhf(v);
#else
    return tanhf(v);
#endif
}

__global__ __launch_bounds__(256)
void narx_wmma_kernel(const float* __restrict__ x,
                      const float* __restrict__ W_in, const float* __restrict__ b_in,
                      const float* __restrict__ W_ih, const float* __restrict__ b_ih,
                      const float* __restrict__ b_hh,
                      const float* __restrict__ W_out, const float* __restrict__ b_out,
                      float* __restrict__ out)
{
    const int lane   = threadIdx.x & 31;
    const int wave   = threadIdx.x >> 5;
    const int col    = lane & 15;          // A row (M) / B,C,D column (N)
    const int hiHalf = (lane >> 4) & 1;    // 0: lanes 0-15, 1: lanes 16-31

    // ---------------- loop-invariant weight fragments ----------------
    // Layer 1 is computed TRANSPOSED:  h^T = W_in @ xt^T
    //   A = W_in tiles (16 h-rows x 32 k), loop-invariant.
    //   A layout: lane row M = col; half p -> K = hiHalf*8 + p (+8 for p>=8)
    HFrag w1[4];
    #pragma unroll
    for (int nb = 0; nb < 4; ++nb) {
        const int h = nb * 16 + col;
        #pragma unroll
        for (int p = 0; p < 16; ++p) {
            const int k = hiHalf * 8 + ((p < 8) ? p : (8 + p));
            w1[nb].h[p] = (_Float16)W_in[h * INF + k];
        }
    }
    // Layer-1 bias, indexed the way the transposed D tile lands in registers:
    // d[nb].f[v] holds h-index nb*16 + v + hiHalf*8
    float bias1v[4][8];
    #pragma unroll
    for (int nb = 0; nb < 4; ++nb)
        #pragma unroll
        for (int v = 0; v < 8; ++v)
            bias1v[nb][v] = b_in[nb * 16 + v + hiHalf * 8];

    // Layer-2 B: W_ih^T as 64x16 tiles -> 4 N-blocks x 2 K-chunks of 32.
    // B layout: lane=col; half p -> K = hiHalf*8 + p (+8 for p>=8)
    HFrag b2[4][2];
    #pragma unroll
    for (int nb = 0; nb < 4; ++nb) {
        const int o = nb * 16 + col;
        #pragma unroll
        for (int f = 0; f < 2; ++f) {
            #pragma unroll
            for (int p = 0; p < 16; ++p) {
                const int k = f * 32 + hiHalf * 8 + ((p < 8) ? p : (8 + p));
                b2[nb][f].h[p] = (_Float16)W_ih[o * HID + k];
            }
        }
    }
    float bias2[4], wout[4];
    #pragma unroll
    for (int nb = 0; nb < 4; ++nb) {
        bias2[nb] = b_ih[nb * 16 + col] + b_hh[nb * 16 + col];
        wout[nb]  = W_out[nb * 16 + col];
    }
    const float bout = b_out[0];

    // Per-lane feature offsets for xt^T in B layout (t-invariant, hoisted):
    // lane col = tile row (g), half p -> feature k = hiHalf*8 + p (+8 for p>=8).
    // Feature k: k<28 -> x[t+off[k/7], g, k%7]; k>=28 -> x[t+off[k-28], g, 7]
    const int offj[4] = {-1, -3, -2, -1};
    int featoff[16];
    #pragma unroll
    for (int p = 0; p < 16; ++p) {
        const int k = hiHalf * 8 + ((p < 8) ? p : (8 + p));
        int j, c;
        if (k < 28) { j = k / 7; c = k % 7; } else { j = k - 28; c = 7; }
        featoff[p] = offj[j] * (NGR * NXC) + c;
    }

    const int nwaves = gridDim.x * 8;
    for (int tl = blockIdx.x * 8 + wave; tl < TILES; tl += nwaves) {
        const int t  = tl / TILES_PER_T + MAXD;
        const int g0 = (tl % TILES_PER_T) * 16;
        const int g  = g0 + col;                  // this lane's tile row

        // ---- build xt^T B fragment (32 k x 16 rows, f16) ----
        const float* xb = x + (size_t)t * (NGR * NXC) + (size_t)g * NXC;
        HFrag bx;
        #pragma unroll
        for (int p = 0; p < 16; ++p)
            bx.h[p] = (_Float16)xb[featoff[p]];

        // ---- layer 1 (transposed): h^T tiles = W_in_tile @ xt^T, 4 WMMAs ----
        FAcc d1[4];
        #pragma unroll
        for (int nb = 0; nb < 4; ++nb) {
            v8f z = {};
            d1[nb].v = __builtin_amdgcn_wmma_f32_16x16x32_f16(
                false, w1[nb].v, false, bx.v, (short)0, z, false, false);
        }

        // ---- bias + relu + pack: d1 registers ARE the layer-2 A fragments ----
        // a2[f].h[p] needs K = f*32 + hiHalf*8 + p (+8 for p>=8)
        //          = h-index held by d1[2f + (p>=8)].f[p%8]
        HFrag a2[2];
        #pragma unroll
        for (int f = 0; f < 2; ++f) {
            #pragma unroll
            for (int p = 0; p < 16; ++p) {
                const int nb = 2 * f + ((p < 8) ? 0 : 1);
                const int v  = (p < 8) ? p : (p - 8);
                float hv = d1[nb].f[v] + bias1v[nb][v];
                hv = hv > 0.0f ? hv : 0.0f;
                a2[f].h[p] = (_Float16)hv;
            }
        }

        // ---- layer 2: a = tanh(h @ W_ih^T + b_ih + b_hh), 8 WMMAs ----
        // ---- layer 3 partials: yacc[r] = sum_nb tanh(..) * W_out ----
        float yacc[8];
        #pragma unroll
        for (int r = 0; r < 8; ++r) yacc[r] = 0.0f;
        #pragma unroll
        for (int nb = 0; nb < 4; ++nb) {
            v8f z = {};
            FAcc acc2;
            acc2.v = __builtin_amdgcn_wmma_f32_16x16x32_f16(
                false, a2[0].v, false, b2[nb][0].v, (short)0, z, false, false);
            acc2.v = __builtin_amdgcn_wmma_f32_16x16x32_f16(
                false, a2[1].v, false, b2[nb][1].v, (short)0, acc2.v, false, false);
            #pragma unroll
            for (int r = 0; r < 8; ++r) {
                const float av = fast_tanh(acc2.f[r] + bias2[nb]);
                yacc[r] += av * wout[nb];
            }
        }

        // ---- reduce across the 16 lanes of each half (cols), store y ----
        #pragma unroll
        for (int r = 0; r < 8; ++r) {
            float s = yacc[r];
            s += __shfl_xor(s, 1, 32);
            s += __shfl_xor(s, 2, 32);
            s += __shfl_xor(s, 4, 32);
            s += __shfl_xor(s, 8, 32);
            if (col == r) {
                const int row = r + hiHalf * 8;
                out[(size_t)t * NGR + g0 + row] = s + bout;
            }
        }
    }
}

// out[t, g] = x[t, g, 7] for t < MAXD
__global__ void narx_head_kernel(const float* __restrict__ x, float* __restrict__ out)
{
    const int i = blockIdx.x * blockDim.x + threadIdx.x;
    if (i < MAXD * NGR)
        out[i] = x[(size_t)i * NXC + 7];
}

extern "C" void kernel_launch(void* const* d_in, const int* in_sizes, int n_in,
                              void* d_out, int out_size, void* d_ws, size_t ws_size,
                              hipStream_t stream)
{
    const float* x     = (const float*)d_in[0];
    const float* W_in  = (const float*)d_in[1];
    const float* b_in  = (const float*)d_in[2];
    const float* W_ih  = (const float*)d_in[3];
    const float* b_ih  = (const float*)d_in[4];
    // d_in[5] = W_hh : dead in the reference forward (only b_hh is used)
    const float* b_hh  = (const float*)d_in[6];
    const float* W_out = (const float*)d_in[7];
    const float* b_out = (const float*)d_in[8];
    float* out = (float*)d_out;

    narx_head_kernel<<<(MAXD * NGR + 255) / 256, 256, 0, stream>>>(x, out);
    narx_wmma_kernel<<<2048, 256, 0, stream>>>(x, W_in, b_in, W_ih, b_ih,
                                               b_hh, W_out, b_out, out);
}
